// EfficientAttention_86895778332971
// MI455X (gfx1250) — matrix operand order repaired and checked by reading, not compiled, not run on hardware
//
#include <hip/hip_runtime.h>
#include <math.h>

// ---------------- CDNA5 WMMA types ----------------
typedef __attribute__((ext_vector_type(16))) __bf16 v16bf;
typedef __attribute__((ext_vector_type(8)))  float  v8f;

// ---------------- Problem constants ----------------
constexpr int NB = 8;        // batch
constexpr int NC = 256;      // channels
constexpr int NS = 16384;    // H*W
constexpr int M3 = 768;      // stacked K/Q/V out channels
constexpr int NH = 2;        // heads
constexpr int HK = 128;      // per-head channels
constexpr float EPS = 1e-5f;

// ---------------- small reduction helper ----------------
__device__ __forceinline__ float block_reduce_sum(float v, float* red) {
  int tid = threadIdx.x;
  red[tid] = v; __syncthreads();
  for (int s = 128; s > 0; s >>= 1) {
    if (tid < s) red[tid] += red[tid + s];
    __syncthreads();
  }
  float r = red[0]; __syncthreads();
  return r;
}
__device__ __forceinline__ float block_reduce_max(float v, float* red) {
  int tid = threadIdx.x;
  red[tid] = v; __syncthreads();
  for (int s = 128; s > 0; s >>= 1) {
    if (tid < s) red[tid] = fmaxf(red[tid], red[tid + s]);
    __syncthreads();
  }
  float r = red[0]; __syncthreads();
  return r;
}

// ---------------- BN stats ----------------
__global__ __launch_bounds__(256) void zero_sums(float* sums) {
  sums[threadIdx.x] = 0.f;
  sums[256 + threadIdx.x] = 0.f;
}

__global__ __launch_bounds__(256) void bn_partial(const float* __restrict__ x,
                                                  float* __restrict__ sums) {
  __shared__ float red[256];
  int c = blockIdx.x, b = blockIdx.y;
  const float* p = x + ((size_t)b * NC + c) * NS;
  float s = 0.f, s2 = 0.f;
  for (int i = threadIdx.x; i < NS; i += 256) {
    float v = p[i]; s += v; s2 += v * v;
  }
  s  = block_reduce_sum(s,  red);
  s2 = block_reduce_sum(s2, red);
  if (threadIdx.x == 0) {
    atomicAdd(&sums[c], s);
    atomicAdd(&sums[256 + c], s2);
  }
}

__global__ __launch_bounds__(256) void finalize_stats(
    const float* __restrict__ sums, const float* __restrict__ gamma,
    const float* __restrict__ beta, const float* __restrict__ bk,
    const float* __restrict__ bq, const float* __restrict__ bv,
    float* __restrict__ ss, float* __restrict__ bcat) {
  int c = threadIdx.x;
  const float invN = 1.0f / ((float)NB * (float)NS);
  float mean = sums[c] * invN;
  float var = sums[256 + c] * invN - mean * mean;
  float rstd = rsqrtf(var + EPS);
  float sc = gamma[c] * rstd;
  ss[c] = sc;
  ss[256 + c] = beta[c] - mean * sc;
  bcat[c] = bk[c];
  bcat[256 + c] = bq[c];
  bcat[512 + c] = bv[c];
}

// ---------------- conversions ----------------
__global__ __launch_bounds__(256) void cvt_x(const float* __restrict__ x,
                                             const float* __restrict__ ss,
                                             __bf16* __restrict__ xn) {
  size_t idx = ((size_t)blockIdx.x * 256 + threadIdx.x) * 4;
  int c = (int)((idx >> 14) & 255);   // (idx / NS) % NC
  float sc = ss[c], sh = ss[256 + c];
  float4 v = *(const float4*)(x + idx);
  xn[idx + 0] = (__bf16)(v.x * sc + sh);
  xn[idx + 1] = (__bf16)(v.y * sc + sh);
  xn[idx + 2] = (__bf16)(v.z * sc + sh);
  xn[idx + 3] = (__bf16)(v.w * sc + sh);
}

__global__ __launch_bounds__(256) void cvt_weights(
    const float* __restrict__ wk, const float* __restrict__ wq,
    const float* __restrict__ wv, const float* __restrict__ wrep_f,
    __bf16* __restrict__ wcat, __bf16* __restrict__ wrep) {
  int idx = (blockIdx.x * 256 + threadIdx.x) * 4;
  #pragma unroll
  for (int j = 0; j < 4; ++j) {
    int id = idx + j;
    if (id < M3 * NC) {
      int o = id >> 8, c = id & 255;
      float v = (o < 256) ? wk[(o << 8) | c]
              : (o < 512) ? wq[((o - 256) << 8) | c]
                          : wv[((o - 512) << 8) | c];
      wcat[id] = (__bf16)v;
    } else {
      int id2 = id - M3 * NC;
      wrep[id2] = (__bf16)wrep_f[id2];
    }
  }
}

// ---------------- generic bf16 WMMA GEMM (async double-buffered) ----------
// C[M,N] (+bias[M]) = A[M,K] * B[K,N], all row-major, A/B bf16, fp32 accum.
// Block tile 128x128, BK=32, 8 waves in 2(M)x4(N), wave tile 64x32 =
// 4x2 v_wmma_f32_16x16x32_bf16 accumulators. Tiles are staged with the CDNA5
// async global->LDS engine (ASYNCcnt) and double-buffered so that the next
// tile's DMA overlaps the current tile's 64 WMMAs.
// Note (ISA 10.2): low 32 bits of a flat LDS address are the LDS byte offset,
// and the async instruction's inst_offset is added to BOTH lds and global
// addresses, so "offset:16" fetches the second 16B half of each 32B chunk.
template <typename OutT>
__global__ __launch_bounds__(256) void gemm_bf16_wmma(
    const __bf16* __restrict__ Ab, const __bf16* __restrict__ Bb,
    const float* __restrict__ bias, OutT* __restrict__ Ob,
    int M, int N, int K, int lda, int ldb, int ldo,
    long long sAo, long long sAi, long long sBo, long long sBi,
    long long sOo, long long sOi, int innerCount) {
  __shared__ alignas(16) __bf16 As[2][128][40];  // 32 + 8 pad (80B row stride)
  __shared__ alignas(16) __bf16 Bs[2][32][136];  // 128 + 8 pad (272B row stride)

  int z = blockIdx.z;
  int outer = z / innerCount, inner = z % innerCount;
  const __bf16* A = Ab + outer * sAo + inner * sAi;
  const __bf16* Bp = Bb + outer * sBo + inner * sBi;
  OutT* O = Ob + outer * sOo + inner * sOi;

  int tid = threadIdx.x;
  int lane = tid & 31, wave = tid >> 5;
  int waveM = wave >> 2, waveN = wave & 3;     // 2 x 4 wave grid
  int m0g = blockIdx.y * 128, n0g = blockIdx.x * 128;
  int rowLane = lane & 15, hi = lane >> 4;

  // per-thread staging coordinates (32B contiguous chunk each)
  int rA = tid >> 1, chA = (tid & 1) * 16;     // A: 128 rows x 32 cols
  int rB = tid >> 3, chB = (tid & 7) * 16;     // B: 32 rows x 128 cols

  v8f acc[4][2];
  #pragma unroll
  for (int mi = 0; mi < 4; ++mi)
    #pragma unroll
    for (int ni = 0; ni < 2; ++ni) { v8f zz = {}; acc[mi][ni] = zz; }

  auto issue_tile = [&](int k0, int buf) {
    unsigned lA = (unsigned)(uintptr_t)&As[buf][rA][chA];
    unsigned long long gA =
        (unsigned long long)(uintptr_t)(A + (size_t)(m0g + rA) * lda + k0 + chA);
    unsigned lB = (unsigned)(uintptr_t)&Bs[buf][rB][chB];
    unsigned long long gB =
        (unsigned long long)(uintptr_t)(Bp + (size_t)(k0 + rB) * ldb + n0g + chB);
    asm volatile(
        "global_load_async_to_lds_b128 %0, %1, off\n\t"
        "global_load_async_to_lds_b128 %0, %1, off offset:16\n\t"
        "global_load_async_to_lds_b128 %2, %3, off\n\t"
        "global_load_async_to_lds_b128 %2, %3, off offset:16"
        :: "v"(lA), "v"(gA), "v"(lB), "v"(gB)
        : "memory");
  };

  issue_tile(0, 0);
  int buf = 0;
  for (int k0 = 0; k0 < K; k0 += 32) {
    asm volatile("s_wait_asynccnt 0x0" ::: "memory");
    __syncthreads();                       // tile[buf] visible to all waves
    if (k0 + 32 < K) issue_tile(k0 + 32, buf ^ 1);

    // A fragment layout (16-bit 16x32): lane holds row M=lane&15,
    // K = ka + i, ka = (lane>=16)?16:0.
    v16bf afrag[4];
    int ka = hi * 16;
    #pragma unroll
    for (int mi = 0; mi < 4; ++mi) {
      int row = waveM * 64 + mi * 16 + rowLane;
      #pragma unroll
      for (int i = 0; i < 8; ++i) afrag[mi][i] = As[buf][row][ka + i];
      #pragma unroll
      for (int i = 8; i < 16; ++i) afrag[mi][i] = As[buf][row][ka + i];
    }
    // B fragment layout (16-bit 32x16): lane holds col N=lane&15,
    // K = kb + i, kb = (lane>=16)?16:0.
    v16bf bfrag[2];
    int kb = hi * 16;
    #pragma unroll
    for (int ni = 0; ni < 2; ++ni) {
      int col = waveN * 32 + ni * 16 + rowLane;
      #pragma unroll
      for (int i = 0; i < 16; ++i) bfrag[ni][i] = Bs[buf][kb + i][col];
    }

    #pragma unroll
    for (int mi = 0; mi < 4; ++mi)
      #pragma unroll
      for (int ni = 0; ni < 2; ++ni)
        acc[mi][ni] = __builtin_amdgcn_wmma_f32_16x16x32_bf16(
            false, afrag[mi], false, bfrag[ni], (short)0, acc[mi][ni],
            false, false);
    __syncthreads();                       // all waves done with tile[buf]
    buf ^= 1;
  }

  // D layout: VGPR r -> lanes 0-15: (M=r, N=lane); lanes 16-31: (M=r+8, N=lane-16)
  #pragma unroll
  for (int mi = 0; mi < 4; ++mi) {
    #pragma unroll
    for (int ni = 0; ni < 2; ++ni) {
      #pragma unroll
      for (int r = 0; r < 8; ++r) {
        int row = m0g + waveM * 64 + mi * 16 + r + (hi << 3);
        int col = n0g + waveN * 32 + ni * 16 + rowLane;
        float v = acc[mi][ni][r];
        if (bias) v += bias[row];
        O[(size_t)row * ldo + col] = (OutT)v;
      }
    }
  }
}

// ---------------- softmaxes ----------------
// K rows: softmax over S (contiguous). One block per (b, channel) row.
__global__ __launch_bounds__(256) void softmax_spatial(__bf16* __restrict__ kqv) {
  __shared__ float red[256];
  int b = blockIdx.x >> 8, r = blockIdx.x & 255;
  __bf16* p = kqv + ((size_t)b * M3 + r) * NS;
  int tid = threadIdx.x;
  float vals[64];
  float mx = -INFINITY;
  #pragma unroll
  for (int i = 0; i < 64; ++i) {
    vals[i] = (float)p[tid + i * 256];
    mx = fmaxf(mx, vals[i]);
  }
  mx = block_reduce_max(mx, red);
  float s = 0.f;
  #pragma unroll
  for (int i = 0; i < 64; ++i) {
    vals[i] = __expf(vals[i] - mx);
    s += vals[i];
  }
  s = block_reduce_sum(s, red);
  float inv = 1.f / s;
  #pragma unroll
  for (int i = 0; i < 64; ++i) p[tid + i * 256] = (__bf16)(vals[i] * inv);
}

// Q: softmax down the 128 head-channels (stride S). One thread per (b,h,s).
// q (64 MiB) is L2-resident (192 MB L2), so the strided passes stay on-chip.
__global__ __launch_bounds__(256) void softmax_channel(__bf16* __restrict__ kqv) {
  int idx = blockIdx.x * 256 + threadIdx.x;      // [0, NB*NH*NS)
  int s = idx & (NS - 1);
  int bh = idx >> 14;
  int b = bh >> 1, h = bh & 1;
  __bf16* p = kqv + ((size_t)b * M3 + 256 + h * HK) * NS + s;
  float m = -INFINITY, sum = 0.f;
  for (int kk = 0; kk < HK; ++kk) {
    float v = (float)p[(size_t)kk * NS];
    float nm = fmaxf(m, v);
    sum = sum * __expf(m - nm) + __expf(v - nm);
    m = nm;
  }
  float inv = 1.f / sum;
  for (int kk = 0; kk < HK; ++kk) {
    float v = (float)p[(size_t)kk * NS];
    p[(size_t)kk * NS] = (__bf16)(__expf(v - m) * inv);
  }
}

// ---------------- context = K @ V^T  (per b,head: 128x128, Kdim=S) ----------
// One wave per 16x16 tile; A(=K rows) and B(=V rows, i.e. V^T cols) fragments
// are contiguous 32B runs in global memory -> direct clause loads, no LDS.
__global__ __launch_bounds__(256) void context_wmma(const __bf16* __restrict__ kqv,
                                                    float* __restrict__ ctx) {
  int bh = blockIdx.x >> 3;
  int tblk = blockIdx.x & 7;
  int wave = threadIdx.x >> 5, lane = threadIdx.x & 31;
  int t = tblk * 8 + wave;             // tile 0..63
  int mi = t >> 3, ni = t & 7;
  int b = bh >> 1, h = bh & 1;
  const __bf16* kp = kqv + ((size_t)b * M3 + h * HK) * NS;
  const __bf16* vp = kqv + ((size_t)b * M3 + 512 + h * HK) * NS;
  int rowLane = lane & 15, hi = lane >> 4;
  const __bf16* arow = kp + (size_t)(mi * 16 + rowLane) * NS;
  const __bf16* brow = vp + (size_t)(ni * 16 + rowLane) * NS;
  v8f acc = {};
  for (int kk = 0; kk < NS; kk += 32) {
    v16bf a, bb;
    int ka = kk + hi * 16;
    #pragma unroll
    for (int i = 0; i < 8; ++i) a[i] = arow[ka + i];
    #pragma unroll
    for (int i = 8; i < 16; ++i) a[i] = arow[ka + i];
    int kb = kk + hi * 16;
    #pragma unroll
    for (int i = 0; i < 16; ++i) bb[i] = brow[kb + i];
    acc = __builtin_amdgcn_wmma_f32_16x16x32_bf16(false, a, false, bb,
                                                  (short)0, acc, false, false);
  }
  float* out = ctx + (size_t)bh * HK * HK;
  #pragma unroll
  for (int r = 0; r < 8; ++r) {
    int row = mi * 16 + r + (hi << 3);
    int col = ni * 16 + rowLane;
    out[row * HK + col] = acc[r];
  }
}

// ---------------- SE gate + head-combine + transpose to bf16 --------------
__global__ __launch_bounds__(256) void se_gate(
    const float* __restrict__ ctx, const float* __restrict__ w_fc1,
    const float* __restrict__ b_fc1, const float* __restrict__ w_fc2,
    const float* __restrict__ b_fc2, const float* __restrict__ w_se,
    const float* __restrict__ b_se, __bf16* __restrict__ ctxT) {
  __shared__ float red[256];
  __shared__ float gateS[2];
  int b = blockIdx.x, tid = threadIdx.x;
  const float* c0 = ctx + (size_t)b * 2 * HK * HK;
  const float* c1 = c0 + HK * HK;
  for (int h = 0; h < 2; ++h) {
    const float* cp = h ? c1 : c0;
    float loc = 0.f;
    for (int i = tid; i < HK * HK; i += 256) loc += cp[i];
    loc = block_reduce_sum(loc, red);
    if (tid == 0) gateS[h] = loc / (float)(HK * HK);
    __syncthreads();
  }
  if (tid == 0) {
    float p0 = gateS[0], p1 = gateS[1];
    float r0 = fmaxf(0.f, p0 * w_fc1[0] + p1 * w_fc1[1] + b_fc1[0]);
    float r1 = fmaxf(0.f, p0 * w_fc1[2] + p1 * w_fc1[3] + b_fc1[1]);
    float z0 = r0 * w_fc2[0] + r1 * w_fc2[1] + b_fc2[0];
    float z1 = r0 * w_fc2[2] + r1 * w_fc2[3] + b_fc2[1];
    gateS[0] = 1.f / (1.f + __expf(-z0));
    gateS[1] = 1.f / (1.f + __expf(-z1));
  }
  __syncthreads();
  float g0 = gateS[0] * w_se[0], g1 = gateS[1] * w_se[1], bs = b_se[0];
  __bf16* outT = ctxT + (size_t)b * HK * HK;
  for (int i = tid; i < HK * HK; i += 256) {
    float v = c0[i] * g0 + c1[i] * g1 + bs;
    int kk = i >> 7, vv = i & 127;
    outT[vv * HK + kk] = (__bf16)v;   // store ctx^T for second-pass GEMM
  }
}

// ---------------- launch ----------------
extern "C" void kernel_launch(void* const* d_in, const int* in_sizes, int n_in,
                              void* d_out, int out_size, void* d_ws,
                              size_t ws_size, hipStream_t stream) {
  const float* x     = (const float*)d_in[0];
  const float* gamma = (const float*)d_in[1];
  const float* beta  = (const float*)d_in[2];
  const float* wk    = (const float*)d_in[3];
  const float* bk    = (const float*)d_in[4];
  const float* wq    = (const float*)d_in[5];
  const float* bq    = (const float*)d_in[6];
  const float* wv    = (const float*)d_in[7];
  const float* bv    = (const float*)d_in[8];
  const float* w_fc1 = (const float*)d_in[9];
  const float* b_fc1 = (const float*)d_in[10];
  const float* w_fc2 = (const float*)d_in[11];
  const float* b_fc2 = (const float*)d_in[12];
  const float* w_se  = (const float*)d_in[13];
  const float* b_se  = (const float*)d_in[14];
  const float* wrepf = (const float*)d_in[15];
  const float* b_rep = (const float*)d_in[16];
  float* out = (float*)d_out;

  char* ws = (char*)d_ws;
  size_t off = 0;
  auto take = [&](size_t bytes) -> char* {
    char* p = ws + off;
    off = (off + bytes + 255) & ~(size_t)255;
    return p;
  };
  float*  d_sums = (float*)take(512 * 4);
  float*  d_ss   = (float*)take(512 * 4);
  float*  d_bcat = (float*)take(768 * 4);
  __bf16* d_wcat = (__bf16*)take((size_t)M3 * NC * 2);
  __bf16* d_wrep = (__bf16*)take((size_t)NC * NC * 2);
  float*  d_ctx  = (float*)take((size_t)NB * NH * HK * HK * 4);
  __bf16* d_ctxT = (__bf16*)take((size_t)NB * HK * HK * 2);
  __bf16* d_xn   = (__bf16*)take((size_t)NB * NC * NS * 2);  // reused as attn out
  __bf16* d_kqv  = (__bf16*)take((size_t)NB * M3 * NS * 2);

  typedef long long ll;

  // 1) BN statistics
  zero_sums<<<1, 256, 0, stream>>>(d_sums);
  bn_partial<<<dim3(NC, NB), 256, 0, stream>>>(x, d_sums);
  finalize_stats<<<1, 256, 0, stream>>>(d_sums, gamma, beta, bk, bq, bv,
                                        d_ss, d_bcat);
  // 2) conversions to bf16
  cvt_weights<<<256, 256, 0, stream>>>(wk, wq, wv, wrepf, d_wcat, d_wrep);
  cvt_x<<<(NB * NC * NS) / 1024, 256, 0, stream>>>(x, d_ss, d_xn);

  // 3) fused K/Q/V GEMM: [768,256] x [256,S] per batch
  gemm_bf16_wmma<__bf16><<<dim3(NS / 128, M3 / 128, NB), 256, 0, stream>>>(
      d_wcat, d_xn, d_bcat, d_kqv, M3, NS, NC, NC, NS, NS,
      (ll)0, (ll)0, (ll)NC * NS, (ll)0, (ll)M3 * NS, (ll)0, 1);

  // 4) softmaxes
  softmax_spatial<<<NB * NC, 256, 0, stream>>>(d_kqv);
  softmax_channel<<<(NB * NH * NS) / 256, 256, 0, stream>>>(d_kqv);

  // 5) context = K @ V^T per (b, head)
  context_wmma<<<NB * NH * 8, 256, 0, stream>>>(d_kqv, d_ctx);

  // 6) SE gate + head combine -> ctx^T (bf16)
  se_gate<<<NB, 256, 0, stream>>>(d_ctx, w_fc1, b_fc1, w_fc2, b_fc2, w_se,
                                  b_se, d_ctxT);

  // 7) attn = ctx^T @ Q per (b, head)  -> reuse xn buffer as bf16 [B,256,S]
  gemm_bf16_wmma<__bf16><<<dim3(NS / 128, 1, NB * NH), 256, 0, stream>>>(
      d_ctxT, d_kqv + (size_t)256 * NS, nullptr, d_xn,
      HK, NS, HK, HK, NS, NS,
      (ll)HK * HK, (ll)0, (ll)M3 * NS, (ll)HK * NS,
      (ll)NC * NS, (ll)HK * NS, NH);

  // 8) out = w_rep @ attn + b_rep  (fp32 to d_out)
  gemm_bf16_wmma<float><<<dim3(NS / 128, NC / 128, NB), 256, 0, stream>>>(
      d_wrep, d_xn, b_rep, out, NC, NS, NC, NC, NS, NS,
      (ll)0, (ll)0, (ll)NC * NS, (ll)0, (ll)NC * NS, (ll)0, 1);
}
